// GraphUNet_35613868819063
// MI455X (gfx1250) — matrix operand order repaired and checked by reading, not compiled, not run on hardware
//
#include <hip/hip_runtime.h>

// ---------------------------------------------------------------------------
// Graph U-Net forward for MI455X (gfx1250, wave32).
// Dense GEMMs via v_wmma_f32_16x16x32_bf16 with fragment-prepacked weights:
//   - A fragment: two global_load_b128 per K-chunk per lane
//   - B fragment: two global_load_b128 per K-chunk per lane (packed layout)
//   - one wave computes a full 16 x cout row-stripe (NT accumulators in VGPRs)
// Edge aggregation via global_atomic_add_f32 (memory-bound, ~8 GB traffic
// -> sub-ms at 23.3 TB/s HBM). Top-k via exact bitonic sort of
// (descending-score, ascending-index) 64-bit keys (deterministic).
// ---------------------------------------------------------------------------

typedef __attribute__((ext_vector_type(16))) __bf16 v16bf;
typedef __attribute__((ext_vector_type(8)))  float  v8f;
typedef unsigned long long u64;
typedef unsigned int u32;

#define cN0  100000
#define cE   1600000
#define cK1  80000
#define cK2  48000
#define cP   131072          /* pow2 >= 100000 for bitonic sort */
#define cEPS 1e-5f

static inline int GRIDN(long n) { return (int)((n + 255) / 256); }

// ----------------------------- utility kernels -----------------------------

__global__ void fill_f32_k(float* __restrict__ p, float v, int n) {
  int i = blockIdx.x * 256 + threadIdx.x;
  if (i < n) p[i] = v;
}

__global__ void fill_i32_k(int* __restrict__ p, int v, int n) {
  int i = blockIdx.x * 256 + threadIdx.x;
  if (i < n) p[i] = v;
}

__global__ void f32_to_bf16_k(const float* __restrict__ in, __bf16* __restrict__ out, int n) {
  int i = blockIdx.x * 256 + threadIdx.x;
  if (i < n) out[i] = (__bf16)in[i];
}

// ------------------------- weight fragment packing -------------------------
// Pack W (f32, cin x cout row-major) into per-lane-contiguous bf16 WMMA B
// fragments: WB[(((kc*nt)+tn)*32 + lane)*16 + t], where lanes 0-15 hold
// K (kc*32 .. +15) of column tn*16+lane, lanes 16-31 hold K (+16 .. +31).

__global__ void pack_w_k(const float* __restrict__ W, __bf16* __restrict__ WB,
                         int cin, int cout) {
  int gid = blockIdx.x * 256 + threadIdx.x;
  if (gid >= cin * cout) return;
  int t    = gid & 15;
  int lane = (gid >> 4) & 31;
  int rest = gid >> 9;                   // kc*nt + tn
  int nt   = cout >> 4;
  int tn   = rest % nt;
  int kc   = rest / nt;
  int col  = (tn << 4) + (lane & 15);
  int k    = (kc << 5) + ((lane < 16) ? 0 : 16) + t;
  WB[gid] = (__bf16)W[(size_t)k * cout + col];
}

// ------------------------------- WMMA GEMM ---------------------------------
// C[n x cout] = A[n x cin] * W[cin x cout]; n%16==0, cin%32==0, cout=NT*16.
// One wave per 16-row stripe; NT accumulators held in registers.

template <int NT>
__global__ void wmma_gemm_bf16_k(const __bf16* __restrict__ A,
                                 const __bf16* __restrict__ WB,
                                 float* __restrict__ C,
                                 int n, int cin) {
  const int cout = NT * 16;
  int wave = blockIdx.x * (blockDim.x >> 5) + (threadIdx.x >> 5);
  int lane = threadIdx.x & 31;
  if (wave >= (n >> 4)) return;          // uniform per wave; EXEC all-1s inside

  int rowA = (wave << 4) + (lane & 15);
  int kloA = (lane < 16) ? 0 : 16;       // A: lanes 0-15 hold K {0..15}

  v8f zero = {};
  v8f acc[NT];
#pragma unroll
  for (int t = 0; t < NT; ++t) acc[t] = zero;

  const int nchunks = cin >> 5;
  for (int kc = 0; kc < nchunks; ++kc) {
    union { uint4 u[2]; v16bf v; } ua;
    const __bf16* ap = A + (size_t)rowA * cin + (kc << 5) + kloA;
    ua.u[0] = *(const uint4*)(ap);       // 16B aligned
    ua.u[1] = *(const uint4*)(ap + 8);
    v16bf a = ua.v;

    const __bf16* bp = WB + (((size_t)kc * NT) * 32 + lane) * 16;
#pragma unroll
    for (int tn = 0; tn < NT; ++tn) {
      union { uint4 u[2]; v16bf v; } ub;
      const __bf16* bf = bp + (size_t)tn * 32 * 16;
      ub.u[0] = *(const uint4*)(bf);     // 32B-aligned contiguous fragment
      ub.u[1] = *(const uint4*)(bf + 8);
      acc[tn] = __builtin_amdgcn_wmma_f32_16x16x32_bf16(false, a, false, ub.v,
                                                        (short)0, acc[tn],
                                                        false, false);
    }
  }

  // D layout: element e -> row e (lanes 0-15) / e+8 (lanes 16-31), col lane&15
  int r0 = (wave << 4) + ((lane < 16) ? 0 : 8);
  int c0 = lane & 15;
#pragma unroll
  for (int tn = 0; tn < NT; ++tn)
#pragma unroll
    for (int e = 0; e < 8; ++e)
      C[(size_t)(r0 + e) * cout + (tn * 16 + c0)] = acc[tn][e];
}

// ------------------------------ GCN pieces ---------------------------------

__global__ void deg_accum_k(const int* __restrict__ dst, const float* __restrict__ ew,
                            float* __restrict__ deg, int E) {
  int e = blockIdx.x * 256 + threadIdx.x;
  if (e < E) atomicAdd(&deg[dst[e]], ew ? ew[e] : 1.0f);
}

__global__ void dinv_k(float* __restrict__ deg, int n) {
  int i = blockIdx.x * 256 + threadIdx.x;
  if (i < n) deg[i] = rsqrtf(deg[i] + 1.0f);   // deg buffer now holds dinv
}

__global__ void agg_k(const int* __restrict__ src, const int* __restrict__ dst,
                      const float* __restrict__ ew, const float* __restrict__ dinv,
                      const float* __restrict__ H, float* __restrict__ AGG,
                      int E, int cout) {
  int gid = blockIdx.x * 256 + threadIdx.x;
  int e = gid / cout;
  if (e >= E) return;
  int c = gid - e * cout;
  float w = ew ? ew[e] : 1.0f;
  if (w == 0.0f) return;
  int s = src[e], d = dst[e];
  float nm = dinv[s] * dinv[d] * w;
  atomicAdd(&AGG[(size_t)d * cout + c], H[(size_t)s * cout + c] * nm);
}

__global__ void finalize_k(float* __restrict__ AGG, const float* __restrict__ H,
                           const float* __restrict__ dinv, const float* __restrict__ b,
                           int n, int cout) {
  int gid = blockIdx.x * 256 + threadIdx.x;
  if (gid >= n * cout) return;
  int i = gid / cout, c = gid - i * cout;
  float dv = dinv[i];                          // rsqrt(deg); h/deg = h*dv*dv
  AGG[gid] = AGG[gid] + H[gid] * dv * dv + b[c];
}

__global__ void bn_stats_k(const float* __restrict__ v, int n, int cout,
                           float* __restrict__ mean, float* __restrict__ var) {
  int c = blockIdx.x;
  __shared__ float s1[256], s2[256];
  float a = 0.f, b = 0.f;
  for (int i = threadIdx.x; i < n; i += 256) {
    float t = v[(size_t)i * cout + c];
    a += t; b += t * t;
  }
  s1[threadIdx.x] = a; s2[threadIdx.x] = b;
  __syncthreads();
  for (int st = 128; st > 0; st >>= 1) {
    if ((int)threadIdx.x < st) { s1[threadIdx.x] += s1[threadIdx.x + st];
                                 s2[threadIdx.x] += s2[threadIdx.x + st]; }
    __syncthreads();
  }
  if (threadIdx.x == 0) {
    float mu = s1[0] / (float)n;
    mean[c] = mu;
    var[c]  = s2[0] / (float)n - mu * mu;      // biased variance
  }
}

__global__ void bn_apply_k(const float* __restrict__ v, const float* __restrict__ mean,
                           const float* __restrict__ var, const float* __restrict__ gamma,
                           const float* __restrict__ beta, float* __restrict__ out,
                           int n, int cout) {
  int gid = blockIdx.x * 256 + threadIdx.x;
  if (gid >= n * cout) return;
  int c = gid % cout;
  float inv = rsqrtf(var[c] + cEPS);
  out[gid] = fmaxf(gamma[c] * (v[gid] - mean[c]) * inv + beta[c], 0.0f);
}

// ----------------------------- embed / output ------------------------------

__global__ void emb_k(const float* __restrict__ x, const float* __restrict__ W,
                      const float* __restrict__ b, float* __restrict__ out, int n) {
  int gid = blockIdx.x * 256 + threadIdx.x;
  if (gid >= n * 32) return;
  int i = gid >> 5, c = gid & 31;
  float s = b[c];
#pragma unroll
  for (int k = 0; k < 5; ++k) s += x[i * 5 + k] * W[k * 32 + c];
  out[gid] = fmaxf(s, 0.0f);
}

__global__ void out_k(const float* __restrict__ h, const float* __restrict__ W,
                      const float* __restrict__ b, float* __restrict__ out, int n) {
  int gid = blockIdx.x * 256 + threadIdx.x;
  if (gid >= n * 2) return;
  int i = gid >> 1, o = gid & 1;
  float s = b[o];
#pragma unroll
  for (int k = 0; k < 32; ++k) s += h[i * 32 + k] * W[k * 2 + o];
  out[gid] = s;
}

// ------------------------------- top-k pool --------------------------------

__global__ void wnorm_k(const float* __restrict__ w, int C, float* __restrict__ out) {
  __shared__ float s[256];
  float a = 0.f;
  for (int i = threadIdx.x; i < C; i += 256) a += w[i] * w[i];
  s[threadIdx.x] = a;
  __syncthreads();
  for (int st = 128; st > 0; st >>= 1) {
    if ((int)threadIdx.x < st) s[threadIdx.x] += s[threadIdx.x + st];
    __syncthreads();
  }
  if (threadIdx.x == 0) out[0] = sqrtf(s[0]);
}

__global__ void score_k(const float* __restrict__ x, const float* __restrict__ w,
                        const float* __restrict__ wn, float* __restrict__ score,
                        int n, int C) {
  int i = blockIdx.x * 256 + threadIdx.x;
  if (i >= n) return;
  float s = 0.f;
  for (int k = 0; k < C; ++k) s += x[(size_t)i * C + k] * w[k];
  score[i] = tanhf(s / wn[0]);
}

__global__ void build_keys_k(const float* __restrict__ score, u64* __restrict__ keys,
                             int n, int P) {
  int i = blockIdx.x * 256 + threadIdx.x;
  if (i >= P) return;
  if (i < n) {
    u32 bits = __float_as_uint(score[i]);
    u32 u = bits ^ ((bits & 0x80000000u) ? 0xFFFFFFFFu : 0x80000000u);  // asc-monotone
    keys[i] = ((u64)(~u) << 32) | (u32)i;      // ascending sort => descending score
  } else {
    keys[i] = ~0ULL;                           // padding sorts last
  }
}

__global__ void bitonic_k(u64* __restrict__ keys, int j, int k, int P) {
  int i = blockIdx.x * 256 + threadIdx.x;
  int ixj = i ^ j;
  if (i < P && ixj > i) {
    u64 a = keys[i], b = keys[ixj];
    bool up = ((i & k) == 0);
    if (up ? (a > b) : (a < b)) { keys[i] = b; keys[ixj] = a; }
  }
}

__global__ void gather_scale_k(const u64* __restrict__ keys, const float* __restrict__ score,
                               const float* __restrict__ xin, float* __restrict__ xout,
                               int k, int C) {
  int gid = blockIdx.x * 256 + threadIdx.x;
  if (gid >= k * C) return;
  int i = gid / C, c = gid - i * C;
  int p = (int)(u32)(keys[i] & 0xFFFFFFFFu);
  xout[gid] = xin[(size_t)p * C + c] * score[p];
}

__global__ void set_newid_k(const u64* __restrict__ keys, int* __restrict__ newid, int k) {
  int i = blockIdx.x * 256 + threadIdx.x;
  if (i < k) newid[(u32)(keys[i] & 0xFFFFFFFFu)] = i;
}

__global__ void remap_k(const int* __restrict__ src, const int* __restrict__ dst,
                        const float* __restrict__ ew, const int* __restrict__ newid,
                        int* __restrict__ so, int* __restrict__ dso,
                        float* __restrict__ ewo, int E) {
  int e = blockIdx.x * 256 + threadIdx.x;
  if (e >= E) return;
  int s2 = newid[src[e]], d2 = newid[dst[e]];
  float w = ew ? ew[e] : 1.0f;
  bool valid = (s2 >= 0) && (d2 >= 0) && (w > 0.5f);
  so[e]  = valid ? s2 : 0;
  dso[e] = valid ? d2 : 0;
  ewo[e] = valid ? 1.0f : 0.0f;
}

// ------------------------------ unpool+concat ------------------------------

__global__ void cat0_k(const float* __restrict__ bt, const float* __restrict__ d0,
                       float* __restrict__ cat, int n) {   // [unpool(bt) 128 | d0 64]
  int gid = blockIdx.x * 256 + threadIdx.x;
  if (gid >= n * 192) return;
  int i = gid / 192, c = gid - i * 192;
  float v;
  if (c < 128) v = (i < cK2) ? bt[(size_t)i * 128 + c] : 0.0f;
  else         v = d0[(size_t)i * 64 + (c - 128)];
  cat[gid] = v;
}

__global__ void cat1_k(const float* __restrict__ h, const float* __restrict__ x0,
                       float* __restrict__ cat, int n) {   // [unpool(h) 64 | x0 32]
  int gid = blockIdx.x * 256 + threadIdx.x;
  if (gid >= n * 96) return;
  int i = gid / 96, c = gid - i * 96;
  float v;
  if (c < 64) v = (i < cK1) ? h[(size_t)i * 64 + c] : 0.0f;
  else        v = x0[(size_t)i * 32 + (c - 64)];
  cat[gid] = v;
}

// --------------------------- host orchestration ----------------------------

static void gcn_block(hipStream_t stream,
                      const float* xin, int n, int cin,
                      const int* src, const int* dst, const float* ew,
                      const float* W, const float* b,
                      const float* gamma, const float* beta,
                      int cout, float* out,
                      __bf16* xb, __bf16* wb, float* H, float* AGG,
                      float* deg, float* stats) {
  f32_to_bf16_k<<<GRIDN((long)n * cin), 256, 0, stream>>>(xin, xb, n * cin);
  pack_w_k<<<GRIDN((long)cin * cout), 256, 0, stream>>>(W, wb, cin, cout);

  int mtiles = n / 16;
  int blocks = (mtiles + 7) / 8;                 // 8 waves / block
  if (cout == 32)
    wmma_gemm_bf16_k<2><<<blocks, 256, 0, stream>>>(xb, wb, H, n, cin);
  else if (cout == 64)
    wmma_gemm_bf16_k<4><<<blocks, 256, 0, stream>>>(xb, wb, H, n, cin);
  else
    wmma_gemm_bf16_k<8><<<blocks, 256, 0, stream>>>(xb, wb, H, n, cin);

  fill_f32_k<<<GRIDN(n), 256, 0, stream>>>(deg, 0.0f, n);
  deg_accum_k<<<GRIDN(cE), 256, 0, stream>>>(dst, ew, deg, cE);
  dinv_k<<<GRIDN(n), 256, 0, stream>>>(deg, n);

  fill_f32_k<<<GRIDN((long)n * cout), 256, 0, stream>>>(AGG, 0.0f, n * cout);
  agg_k<<<GRIDN((long)cE * cout), 256, 0, stream>>>(src, dst, ew, deg, H, AGG, cE, cout);

  finalize_k<<<GRIDN((long)n * cout), 256, 0, stream>>>(AGG, H, deg, b, n, cout);
  bn_stats_k<<<cout, 256, 0, stream>>>(AGG, n, cout, stats, stats + 256);
  bn_apply_k<<<GRIDN((long)n * cout), 256, 0, stream>>>(AGG, stats, stats + 256,
                                                        gamma, beta, out, n, cout);
}

static void topk_pool(hipStream_t stream,
                      const float* xin, int nprev, int C, const float* w,
                      const int* src, const int* dst, const float* ew, int k,
                      float* xout, int* so, int* dso, float* ewo,
                      float* score, u64* keys, int* newid, float* wn) {
  wnorm_k<<<1, 256, 0, stream>>>(w, C, wn);
  score_k<<<GRIDN(nprev), 256, 0, stream>>>(xin, w, wn, score, nprev, C);
  build_keys_k<<<GRIDN(cP), 256, 0, stream>>>(score, keys, nprev, cP);
  for (int kk = 2; kk <= cP; kk <<= 1)
    for (int j = kk >> 1; j > 0; j >>= 1)
      bitonic_k<<<cP / 256, 256, 0, stream>>>(keys, j, kk, cP);
  gather_scale_k<<<GRIDN((long)k * C), 256, 0, stream>>>(keys, score, xin, xout, k, C);
  fill_i32_k<<<GRIDN(nprev), 256, 0, stream>>>(newid, -1, nprev);
  set_newid_k<<<GRIDN(k), 256, 0, stream>>>(keys, newid, k);
  remap_k<<<GRIDN(cE), 256, 0, stream>>>(src, dst, ew, newid, so, dso, ewo, cE);
}

extern "C" void kernel_launch(void* const* d_in, const int* in_sizes, int n_in,
                              void* d_out, int out_size, void* d_ws, size_t ws_size,
                              hipStream_t stream) {
  (void)in_sizes; (void)n_in; (void)out_size; (void)ws_size;

  // --- inputs (setup_inputs() insertion order, params flattened depth-first) ---
  const float* x     = (const float*)d_in[0];
  const int*   ei    = (const int*)d_in[1];
  const float* emb_w = (const float*)d_in[2];
  const float* emb_b = (const float*)d_in[3];
  const float* d0W = (const float*)d_in[4],  *d0b = (const float*)d_in[5];
  const float* d0g = (const float*)d_in[6],  *d0be = (const float*)d_in[7];
  const float* d1W = (const float*)d_in[8],  *d1b = (const float*)d_in[9];
  const float* d1g = (const float*)d_in[10], *d1be = (const float*)d_in[11];
  const float* pw0 = (const float*)d_in[12];
  const float* pw1 = (const float*)d_in[13];
  const float* btW = (const float*)d_in[14], *btb = (const float*)d_in[15];
  const float* btg = (const float*)d_in[16], *btbe = (const float*)d_in[17];
  const float* u0W = (const float*)d_in[18], *u0b = (const float*)d_in[19];
  const float* u0g = (const float*)d_in[20], *u0be = (const float*)d_in[21];
  const float* u1W = (const float*)d_in[22], *u1b = (const float*)d_in[23];
  const float* u1g = (const float*)d_in[24], *u1be = (const float*)d_in[25];
  const float* oW  = (const float*)d_in[26], *ob = (const float*)d_in[27];

  const int* src0 = ei;
  const int* dst0 = ei + cE;

  // --- workspace arena ---
  char* base = (char*)d_ws;
  size_t off = 0;
  auto alloc = [&](size_t bytes) -> void* {
    off = (off + 255) & ~(size_t)255;
    void* p = base + off;
    off += bytes;
    return p;
  };

  __bf16* xb16  = (__bf16*)alloc((size_t)cN0 * 192 * 2);
  __bf16* wb16  = (__bf16*)alloc((size_t)192 * 128 * 2);
  float*  H     = (float*)alloc((size_t)cN0 * 128 * 4);
  float*  AGG   = (float*)alloc((size_t)cN0 * 128 * 4);
  float*  x0    = (float*)alloc((size_t)cN0 * 32 * 4);
  float*  d0    = (float*)alloc((size_t)cN0 * 64 * 4);
  float*  x1    = (float*)alloc((size_t)cK1 * 64 * 4);
  float*  d1    = (float*)alloc((size_t)cK1 * 128 * 4);
  float*  x2    = (float*)alloc((size_t)cK2 * 128 * 4);
  float*  bt    = (float*)alloc((size_t)cK2 * 128 * 4);
  float*  cat   = (float*)alloc((size_t)cN0 * 192 * 4);   // reused for 96-wide concat
  float*  hup   = (float*)alloc((size_t)cN0 * 64 * 4);
  float*  hup2  = (float*)alloc((size_t)cN0 * 32 * 4);
  float*  deg   = (float*)alloc((size_t)cN0 * 4);
  float*  score = (float*)alloc((size_t)cN0 * 4);
  u64*    keys  = (u64*)alloc((size_t)cP * 8);
  int*    newid = (int*)alloc((size_t)cN0 * 4);
  int*    src1  = (int*)alloc((size_t)cE * 4);
  int*    dst1  = (int*)alloc((size_t)cE * 4);
  float*  ew1   = (float*)alloc((size_t)cE * 4);
  int*    src2  = (int*)alloc((size_t)cE * 4);
  int*    dst2  = (int*)alloc((size_t)cE * 4);
  float*  ew2   = (float*)alloc((size_t)cE * 4);
  float*  stats = (float*)alloc(512 * 4);                  // mean[256] | var[256]
  float*  wn    = (float*)alloc(256);

  // --- pipeline ---
  emb_k<<<GRIDN((long)cN0 * 32), 256, 0, stream>>>(x, emb_w, emb_b, x0, cN0);

  gcn_block(stream, x0, cN0, 32, src0, dst0, nullptr,
            d0W, d0b, d0g, d0be, 64, d0, xb16, wb16, H, AGG, deg, stats);

  topk_pool(stream, d0, cN0, 64, pw0, src0, dst0, nullptr, cK1,
            x1, src1, dst1, ew1, score, keys, newid, wn);

  gcn_block(stream, x1, cK1, 64, src1, dst1, ew1,
            d1W, d1b, d1g, d1be, 128, d1, xb16, wb16, H, AGG, deg, stats);

  topk_pool(stream, d1, cK1, 128, pw1, src1, dst1, ew1, cK2,
            x2, src2, dst2, ew2, score, keys, newid, wn);

  gcn_block(stream, x2, cK2, 128, src2, dst2, ew2,
            btW, btb, btg, btbe, 128, bt, xb16, wb16, H, AGG, deg, stats);

  cat0_k<<<GRIDN((long)cN0 * 192), 256, 0, stream>>>(bt, d0, cat, cN0);
  gcn_block(stream, cat, cN0, 192, src1, dst1, ew1,
            u0W, u0b, u0g, u0be, 64, hup, xb16, wb16, H, AGG, deg, stats);

  cat1_k<<<GRIDN((long)cN0 * 96), 256, 0, stream>>>(hup, x0, cat, cN0);
  gcn_block(stream, cat, cN0, 96, src0, dst0, nullptr,
            u1W, u1b, u1g, u1be, 32, hup2, xb16, wb16, H, AGG, deg, stats);

  out_k<<<GRIDN((long)cN0 * 2), 256, 0, stream>>>(hup2, oW, ob, (float*)d_out, cN0);
}
